// Model3_25125558681689
// MI455X (gfx1250) — compile-verified, matrix operand order checked by reading, not run on hardware
//
#include <hip/hip_runtime.h>
#include <hip/hip_bf16.h>
#include <cmath>

// ---------------------------------------------------------------------------
// MI455X (gfx1250) implementation of the point-transformer reference.
// All dense math routed through v_wmma_f32_16x16x32_f16 (f16 in, f32 acc).
// GEMM invariants: K % 32 == 0 (embedding K=3 zero-padded to 32 in ws);
// M,N >= 4; OOB tile rows/cols handled by index clamping (loads stay
// unconditional -> batched issue, single wait) + bounds-checked stores.
//
// Input-order assumption (compile-only): JAX sorted-key pytree flatten:
//   params leaves first ('params' < 'x'), classifier=0..5, embedding=6..11,
//   stages[s]{pos0,pos1,pre0,pre1} with block leaves sorted
//   {fb,fbb,fg,fw,ob,og,wk,wo,wq,wv}; stage base = 12+40*s; x found by size.
// ---------------------------------------------------------------------------

typedef __attribute__((ext_vector_type(16))) _Float16 v16h;
typedef __attribute__((ext_vector_type(8)))  _Float16 v8h;
typedef __attribute__((ext_vector_type(4)))  _Float16 v4h;
typedef __attribute__((ext_vector_type(8)))  float    v8f;

// ============================== WMMA GEMM ==================================
// C[M,N](+bias)(ReLU) = A[M,K] x op(B);  transB=1: B=[N,K]; transB=0: B=[K,N]
// Batched over blockIdx.z (element strides sA,sB,sC).
// Block: 128 threads = 4 wave32; tile 64(M) x 32(N); K-step 32.
// Each wave: one A fragment, two B fragments -> two v_wmma per K-step.
__global__ __launch_bounds__(128) void k_gemm(
    const float* __restrict__ A, const float* __restrict__ B,
    const float* __restrict__ bias, float* __restrict__ C,
    int M, int N, int K, int lda, int ldb, int ldc,
    long long sA, long long sB, long long sC,
    int transB, int relu)
{
    __shared__ __attribute__((aligned(16))) _Float16 Ash[64][40]; // padded stride
    __shared__ __attribute__((aligned(16))) _Float16 Bsh[32][40]; // [n][k] (B^T)

    const int tid  = threadIdx.x;
    const int wid  = tid >> 5;
    const int lane = tid & 31;
    const int half = lane >> 4;
    const int l16  = lane & 15;

    const long long z = blockIdx.z;
    const float* Ab = A + z * sA;
    const float* Bb = B + z * sB;
    float*       Cb = C + z * sC;

    const int m0 = blockIdx.y * 64;
    const int n0 = blockIdx.x * 32;

    v8f acc0 = {};
    v8f acc1 = {};

    for (int k0 = 0; k0 < K; k0 += 32) {
        __syncthreads();
        // ---- stage A tile 64x32: 512 float4-chunks, 4 per thread ----
        // OOB rows clamped to M-1 (their products land only in C rows >= M,
        // which are never stored) -> unconditional, batchable loads.
        float4 va[4];
#pragma unroll
        for (int i = 0; i < 4; ++i) {
            int ch = tid + 128 * i;            // 0..511
            int mm = ch >> 3;                  // 0..63
            int kc = (ch & 7) * 4;             // 0..28
            int gm = m0 + mm; if (gm >= M) gm = M - 1;
            va[i] = *(const float4*)(Ab + (long long)gm * lda + k0 + kc);
        }
#pragma unroll
        for (int i = 0; i < 4; ++i) {
            int ch = tid + 128 * i;
            int mm = ch >> 3;
            int kc = (ch & 7) * 4;
            v4h h = { (_Float16)va[i].x, (_Float16)va[i].y,
                      (_Float16)va[i].z, (_Float16)va[i].w };
            *(v4h*)&Ash[mm][kc] = h;
        }
        // ---- stage B tile 32(n) x 32(k): 256 chunks, 2 per thread ----
        if (transB) {
            float4 vb[2];
#pragma unroll
            for (int i = 0; i < 2; ++i) {
                int ch = tid + 128 * i;        // 0..255
                int nn = ch >> 3;              // 0..31
                int kc = (ch & 7) * 4;
                int gn = n0 + nn; if (gn >= N) gn = N - 1;   // OOB col clamp
                vb[i] = *(const float4*)(Bb + (long long)gn * ldb + k0 + kc);
            }
#pragma unroll
            for (int i = 0; i < 2; ++i) {
                int ch = tid + 128 * i;
                int nn = ch >> 3;
                int kc = (ch & 7) * 4;
                v4h h = { (_Float16)vb[i].x, (_Float16)vb[i].y,
                          (_Float16)vb[i].z, (_Float16)vb[i].w };
                *(v4h*)&Bsh[nn][kc] = h;
            }
        } else {
            // NN path (attn @ V): N = dh is a multiple of 32 here, chunk-base
            // clamp keeps loads unconditional.
            float4 vb[2];
#pragma unroll
            for (int i = 0; i < 2; ++i) {
                int ch = tid + 128 * i;
                int kk = ch >> 3;              // 0..31
                int nc = (ch & 7) * 4;         // 0..28
                int gk = k0 + kk;
                int gn0 = n0 + nc; if (gn0 > N - 4) gn0 = N - 4;
                vb[i] = *(const float4*)(Bb + (long long)gk * ldb + gn0);
            }
#pragma unroll
            for (int i = 0; i < 2; ++i) {
                int ch = tid + 128 * i;
                int kk = ch >> 3;
                int nc = (ch & 7) * 4;
                Bsh[nc + 0][kk] = (_Float16)vb[i].x;
                Bsh[nc + 1][kk] = (_Float16)vb[i].y;
                Bsh[nc + 2][kk] = (_Float16)vb[i].z;
                Bsh[nc + 3][kk] = (_Float16)vb[i].w;
            }
        }
        // prefetch next A K-tile into L2 while we compute
        if (k0 + 32 < K) {
            int pm = m0 + (tid >> 1); if (pm >= M) pm = M - 1;
            __builtin_prefetch(&Ab[(long long)pm * lda + k0 + 32], 0, 1);
        }
        __syncthreads();

        // ---- fragments per ISA 7.12.2: two 16B LDS reads per fragment ----
        const int arow = (wid << 4) + l16;
        v8h alo = *(const v8h*)&Ash[arow][half * 8];
        v8h ahi = *(const v8h*)&Ash[arow][16 + half * 8];
        v8h b0l = *(const v8h*)&Bsh[l16][half * 8];
        v8h b0h = *(const v8h*)&Bsh[l16][16 + half * 8];
        v8h b1l = *(const v8h*)&Bsh[16 + l16][half * 8];
        v8h b1h = *(const v8h*)&Bsh[16 + l16][16 + half * 8];
        v16h af, bf0, bf1;
#pragma unroll
        for (int i = 0; i < 8; ++i) {
            af[i]  = alo[i]; af[i + 8]  = ahi[i];
            bf0[i] = b0l[i]; bf0[i + 8] = b0h[i];
            bf1[i] = b1l[i]; bf1[i + 8] = b1h[i];
        }
        acc0 = __builtin_amdgcn_wmma_f32_16x16x32_f16(
            false, af, false, bf0, (short)0, acc0, false, false);
        acc1 = __builtin_amdgcn_wmma_f32_16x16x32_f16(
            false, af, false, bf1, (short)0, acc1, false, false);
    }

    // ---- epilogue: C layout row = r + 8*half, col = lane%16 ----
    v8f accs[2] = { acc0, acc1 };
#pragma unroll
    for (int t = 0; t < 2; ++t) {
        int col = n0 + t * 16 + l16;
        if (col < N) {
            float bv = bias ? bias[col] : 0.0f;
#pragma unroll
            for (int r = 0; r < 8; ++r) {
                int row = m0 + (wid << 4) + half * 8 + r;
                if (row < M) {
                    float v = accs[t][r] + bv;
                    if (relu) v = fmaxf(v, 0.0f);
                    Cb[(long long)row * ldc + col] = v;
                }
            }
        }
    }
}

// ============================ elementwise kernels ==========================

// x [B,3,N] -> xyz [B*N,3] and zero-padded tokens [B*N,32]
__global__ void k_transpose_pad(const float* __restrict__ x, float* __restrict__ xyz,
                                float* __restrict__ tok, int B, int N)
{
    long long t = (long long)blockIdx.x * blockDim.x + threadIdx.x;
    long long total = (long long)B * N * 32;
    if (t >= total) return;
    int c = (int)(t % 32);
    long long rem = t / 32;
    int n = (int)(rem % N);
    int b = (int)(rem / N);
    float v = 0.0f;
    if (c < 3) {
        v = x[((long long)b * 3 + c) * N + n];
        xyz[rem * 3 + c] = v;
    }
    tok[t] = v;
}

// w [O,3] -> wp [O,32] zero-padded
__global__ void k_pad_w(const float* __restrict__ w, float* __restrict__ wp, int O)
{
    int t = blockIdx.x * blockDim.x + threadIdx.x;
    if (t >= O * 32) return;
    int c = t % 32, o = t / 32;
    wp[t] = (c < 3) ? w[o * 3 + c] : 0.0f;
}

__global__ void k_bn_stats(const float* __restrict__ X, float* __restrict__ st,
                           long long M, int C)
{
    int c = blockIdx.x * 256 + threadIdx.x;
    if (c >= C) return;
    long long r0 = (long long)blockIdx.y * 256;
    long long r1 = r0 + 256; if (r1 > M) r1 = M;
    float s = 0.0f, s2 = 0.0f;
    for (long long r = r0; r < r1; ++r) {
        float v = X[r * C + c];
        s += v; s2 += v * v;
    }
    atomicAdd(&st[c], s);
    atomicAdd(&st[C + c], s2);
}

__global__ void k_bn_apply(const float* __restrict__ X, const float* __restrict__ st,
                           const float* __restrict__ g, const float* __restrict__ b,
                           const float* __restrict__ res, float* __restrict__ out,
                           long long M, int C, int relu)
{
    long long t = (long long)blockIdx.x * blockDim.x + threadIdx.x;
    if (t >= M * (long long)C) return;
    int c = (int)(t % C);
    float mean = st[c] / (float)M;
    float var  = st[C + c] / (float)M - mean * mean;
    float v = (X[t] - mean) * rsqrtf(var + 1e-5f) * g[c] + b[c];
    if (relu) v = fmaxf(v, 0.0f);
    if (res) v += res[t];
    out[t] = v;
}

// wave32-per-row softmax with scale; row length n <= 512
__global__ __launch_bounds__(128) void k_softmax(float* __restrict__ p,
                                                 long long rows, int n, float scale)
{
    int lane = threadIdx.x & 31;
    long long row = (long long)blockIdx.x * 4 + (threadIdx.x >> 5);
    if (row >= rows) return;
    float* x = p + row * (long long)n;
    float mx = -3.4e38f;
    for (int j = lane; j < n; j += 32) mx = fmaxf(mx, x[j] * scale);
    for (int m = 1; m < 32; m <<= 1) mx = fmaxf(mx, __shfl_xor(mx, m, 32));
    float ev[16];
    float sum = 0.0f;
    int i = 0;
    for (int j = lane; j < n; j += 32, ++i) {
        float e = __expf(x[j] * scale - mx);
        ev[i] = e; sum += e;
    }
    for (int m = 1; m < 32; m <<= 1) sum += __shfl_xor(sum, m, 32);
    float inv = 1.0f / sum;
    i = 0;
    for (int j = lane; j < n; j += 32, ++i) x[j] = ev[i] * inv;
}

// O planes [h][M][dh] -> Xc [M][4*dh]
__global__ void k_concat_heads(const float* __restrict__ O, float* __restrict__ Xc,
                               long long Mtot, int dh)
{
    long long t = (long long)blockIdx.x * blockDim.x + threadIdx.x;
    int inner = 4 * dh;
    long long total = Mtot * (long long)inner;
    if (t >= total) return;
    long long m = t / inner;
    int c = (int)(t % inner);
    int h = c / dh, e = c % dh;
    Xc[t] = O[((long long)h * Mtot + m) * dh + e];
}

__global__ void k_maxpool(const float* __restrict__ X, float* __restrict__ Y,
                          long long groups, int Kc, int C)
{
    long long t = (long long)blockIdx.x * blockDim.x + threadIdx.x;
    if (t >= groups * (long long)C) return;
    long long g = t / C;
    int c = (int)(t % C);
    const float* p = X + g * (long long)Kc * C + c;
    float m = p[0];
    for (int i = 1; i < Kc; ++i) m = fmaxf(m, p[(long long)i * C]);
    Y[g * C + c] = m;
}

// ============================ point ops ====================================

__global__ __launch_bounds__(256) void k_fps(const float* __restrict__ xyz,
                                             int* __restrict__ out, int N, int S)
{
    int b = blockIdx.x, tid = threadIdx.x;
    __shared__ float dist[1024];
    __shared__ float rv[256];
    __shared__ int   ri[256];
    __shared__ float fpt[3];
    __shared__ int far_s;
    const float* P = xyz + (long long)b * N * 3;
    for (int i = tid; i < N; i += 256) dist[i] = 1e10f;
    if (tid == 0) far_s = 0;
    __syncthreads();
    for (int s = 0; s < S; ++s) {
        int far = far_s;
        if (tid == 0) {
            out[b * S + s] = far;
            fpt[0] = P[far * 3 + 0]; fpt[1] = P[far * 3 + 1]; fpt[2] = P[far * 3 + 2];
        }
        __syncthreads();
        float bestv = -1.0f; int besti = 0x7fffffff;
        for (int i = tid; i < N; i += 256) {
            float dx = P[i * 3 + 0] - fpt[0];
            float dy = P[i * 3 + 1] - fpt[1];
            float dz = P[i * 3 + 2] - fpt[2];
            float d = dx * dx + dy * dy + dz * dz;
            float nd = fminf(dist[i], d);
            dist[i] = nd;
            if (nd > bestv || (nd == bestv && i < besti)) { bestv = nd; besti = i; }
        }
        rv[tid] = bestv; ri[tid] = besti;
        __syncthreads();
        for (int off = 128; off > 0; off >>= 1) {
            if (tid < off) {
                if (rv[tid + off] > rv[tid] ||
                    (rv[tid + off] == rv[tid] && ri[tid + off] < ri[tid])) {
                    rv[tid] = rv[tid + off]; ri[tid] = ri[tid + off];
                }
            }
            __syncthreads();
        }
        if (tid == 0) far_s = ri[0];
        __syncthreads();
    }
}

__global__ void k_gather_xyz(const float* __restrict__ xyz, const int* __restrict__ fps,
                             float* __restrict__ nxyz, int B, int Nin, int S)
{
    long long t = (long long)blockIdx.x * blockDim.x + threadIdx.x;
    if (t >= (long long)B * S * 3) return;
    int c = (int)(t % 3);
    long long rem = t / 3;
    int s = (int)(rem % S);
    int b = (int)(rem / S);
    int idx = fps[b * S + s];
    nxyz[t] = xyz[((long long)b * Nin + idx) * 3 + c];
}

__global__ void k_knn(const float* __restrict__ xyz, const float* __restrict__ nxyz,
                      int* __restrict__ knn, int B, int N, int S)
{
    int t = blockIdx.x * blockDim.x + threadIdx.x;
    if (t >= B * S) return;
    int b = t / S;
    const float* P = xyz + (long long)b * N * 3;
    float ax = nxyz[t * 3 + 0], ay = nxyz[t * 3 + 1], az = nxyz[t * 3 + 2];
    float bd[32]; int bi[32];
    for (int k = 0; k < 32; ++k) { bd[k] = 3.4e38f; bi[k] = 0; }
    for (int n = 0; n < N; ++n) {
        float dx = P[n * 3 + 0] - ax, dy = P[n * 3 + 1] - ay, dz = P[n * 3 + 2] - az;
        float d = dx * dx + dy * dy + dz * dz;
        if (d < bd[31]) {
            int j = 31;
            while (j > 0 && bd[j - 1] > d) { bd[j] = bd[j - 1]; bi[j] = bi[j - 1]; --j; }
            bd[j] = d; bi[j] = n;
        }
    }
    for (int k = 0; k < 32; ++k) knn[t * 32 + k] = bi[k];
}

__global__ void k_group(const float* __restrict__ feats, const int* __restrict__ fps,
                        const int* __restrict__ knn, float* __restrict__ Y,
                        int B, int Nin, int S, int din)
{
    long long t = (long long)blockIdx.x * blockDim.x + threadIdx.x;
    long long total = (long long)B * S * 32 * din;
    if (t >= total) return;
    int c = (int)(t % din);
    long long rem = t / din;
    int kk = (int)(rem % 32); rem /= 32;
    int s  = (int)(rem % S);
    int b  = (int)(rem / S);
    int anchor = b * S + s;
    int ctr_i = fps[anchor];
    int nb_i  = knn[anchor * 32 + kk];
    float ctr = feats[((long long)b * Nin + ctr_i) * din + c];
    float val = feats[((long long)b * Nin + nb_i) * din + c];
    long long tok = (long long)anchor * 32 + kk;
    Y[tok * (2 * din) + c] = val - ctr;
    Y[tok * (2 * din) + din + c] = ctr;
}

// ============================ host orchestration ===========================

struct BlockP {
    const float *fb, *fbb, *fg, *fw, *ob, *og, *wk, *wo, *wq, *wv;
};

static BlockP get_block(void* const* d_in, int base)
{
    BlockP p;
    p.fb  = (const float*)d_in[base + 0];
    p.fbb = (const float*)d_in[base + 1];
    p.fg  = (const float*)d_in[base + 2];
    p.fw  = (const float*)d_in[base + 3];
    p.ob  = (const float*)d_in[base + 4];
    p.og  = (const float*)d_in[base + 5];
    p.wk  = (const float*)d_in[base + 6];
    p.wo  = (const float*)d_in[base + 7];
    p.wq  = (const float*)d_in[base + 8];
    p.wv  = (const float*)d_in[base + 9];
    return p;
}

static void run_gemm(hipStream_t st, const float* A, const float* B, const float* bias,
                     float* C, int M, int N, int K, int lda, int ldb, int ldc,
                     long long sA, long long sB, long long sC, int batch,
                     int transB, int relu)
{
    dim3 g((N + 31) / 32, (M + 63) / 64, batch);
    k_gemm<<<g, 128, 0, st>>>(A, B, bias, C, M, N, K, lda, ldb, ldc,
                              sA, sB, sC, transB, relu);
}

static void run_bn(hipStream_t st, const float* X, float* stats, const float* g,
                   const float* b, const float* res, float* out,
                   long long M, int C, int relu)
{
    hipMemsetAsync(stats, 0, 2 * (size_t)C * sizeof(float), st);
    dim3 gs((C + 255) / 256, (unsigned)((M + 255) / 256), 1);
    k_bn_stats<<<gs, 256, 0, st>>>(X, stats, M, C);
    long long tot = M * (long long)C;
    k_bn_apply<<<(unsigned)((tot + 255) / 256), 256, 0, st>>>(X, stats, g, b, res, out,
                                                              M, C, relu);
}

// x [G*n, d] updated in place through one transformer block
static void run_block(hipStream_t st, float* x, long long Gg, int n, int d,
                      const BlockP& p, float* q, float* kb, float* v, float* o,
                      float* scores, float* proj, float* att, float* stats)
{
    const long long M = Gg * (long long)n;
    const int dh = d / 2, inner = 2 * d, dg = d / 4;

    // grouped QKV: batch over 4 heads; A head-offset = dg channels within row
    run_gemm(st, x, p.wq, nullptr, q, (int)M, dh, dg, d, dg, dh,
             (long long)dg, (long long)dh * dg, M * (long long)dh, 4, 1, 0);
    run_gemm(st, x, p.wk, nullptr, kb, (int)M, dh, dg, d, dg, dh,
             (long long)dg, (long long)dh * dg, M * (long long)dh, 4, 1, 0);
    run_gemm(st, x, p.wv, nullptr, v, (int)M, dh, dg, d, dg, dh,
             (long long)dg, (long long)dh * dg, M * (long long)dh, 4, 1, 0);

    // scores[bz] = q[bz] @ k[bz]^T, bz = h*G+g
    const int GB = (int)(4 * Gg);
    run_gemm(st, q, kb, nullptr, scores, n, n, dh, dh, dh, n,
             (long long)n * dh, (long long)n * dh, (long long)n * n, GB, 1, 0);

    long long rows = (long long)GB * n;
    float scale = 1.0f / sqrtf((float)dh);
    k_softmax<<<(unsigned)((rows + 3) / 4), 128, 0, st>>>(scores, rows, n, scale);

    // o[bz] = attn[bz] @ v[bz]   (NN)
    run_gemm(st, scores, v, nullptr, o, n, dh, n, n, dh, dh,
             (long long)n * n, (long long)n * dh, (long long)n * dh, GB, 0, 0);

    // concat heads (reuse kb as [M, inner])
    float* xc = kb;
    long long tot = M * (long long)inner;
    k_concat_heads<<<(unsigned)((tot + 255) / 256), 256, 0, st>>>(o, xc, M, dh);

    // out projection + BN + ReLU + residual(x) -> att
    run_gemm(st, xc, p.wo, nullptr, proj, (int)M, d, inner, inner, inner, d,
             0, 0, 0, 1, 1, 0);
    run_bn(st, proj, stats, p.og, p.ob, x, att, M, d, 1);

    // FFN + bias + BN + ReLU + residual(att) -> x (in place)
    run_gemm(st, att, p.fw, p.fb, proj, (int)M, d, d, d, d, d, 0, 0, 0, 1, 1, 0);
    run_bn(st, proj, stats, p.fg, p.fbb, att, x, M, d, 1);
}

extern "C" void kernel_launch(void* const* d_in, const int* in_sizes, int n_in,
                              void* d_out, int out_size, void* d_ws, size_t ws_size,
                              hipStream_t stream)
{
    (void)out_size; (void)ws_size;
    const int B = 16;

    // locate x by its unique flat size; fall back to pytree position 172
    int xi = (n_in > 172) ? 172 : (n_in - 1);
    for (int i = 0; i < n_in; ++i)
        if (in_sizes[i] == 16 * 3 * 1024) { xi = i; break; }
    const float* x_in = (const float*)d_in[xi];

    const float* cls_b  = (const float*)d_in[0];
    const float* cls_b1 = (const float*)d_in[1];
    const float* cls_b2 = (const float*)d_in[2];
    const float* cls_g  = (const float*)d_in[3];
    const float* cls_w1 = (const float*)d_in[4];
    const float* cls_w2 = (const float*)d_in[5];
    const float* emb_b0 = (const float*)d_in[6];
    const float* emb_g0 = (const float*)d_in[7];
    const float* emb_w0 = (const float*)d_in[8];
    const float* emb_b1 = (const float*)d_in[9];
    const float* emb_g1 = (const float*)d_in[10];
    const float* emb_w1 = (const float*)d_in[11];

    // ---- workspace bump allocator ----
    char* base = (char*)d_ws;
    size_t off = 0;
    auto alloc = [&](size_t bytes) -> void* {
        size_t a = (off + 255) & ~(size_t)255;
        off = a + bytes;
        return (void*)(base + a);
    };
    const size_t F = sizeof(float);
    float* xyzA   = (float*)alloc((size_t)16 * 1024 * 3 * F);
    float* xyzB   = (float*)alloc((size_t)16 * 512 * 3 * F);
    float* featsA = (float*)alloc((size_t)1048576 * F);
    float* featsB = (float*)alloc((size_t)1048576 * F);
    int*   fpsIdx = (int*)alloc((size_t)16 * 512 * 4);
    int*   knnIdx = (int*)alloc((size_t)16 * 512 * 32 * 4);
    float* embIn  = (float*)alloc((size_t)16 * 1024 * 32 * F);  // K-padded tokens
    float* embW0p = (float*)alloc((size_t)64 * 32 * F);         // K-padded weight
    float* Y      = (float*)alloc((size_t)33554432 * F);
    float* q      = (float*)alloc((size_t)67108864 * F);
    float* kb     = (float*)alloc((size_t)67108864 * F);   // also concat buffer
    float* v      = (float*)alloc((size_t)67108864 * F);
    float* o      = (float*)alloc((size_t)67108864 * F);
    float* scores = (float*)alloc((size_t)33554432 * F);
    float* proj   = (float*)alloc((size_t)33554432 * F);
    float* att    = (float*)alloc((size_t)33554432 * F);
    float* stats  = (float*)alloc((size_t)2048 * F);
    float* pooled = (float*)alloc((size_t)16 * 1024 * F);
    float* z1     = (float*)alloc((size_t)16 * 256 * F);

    // ---- embedding (K padded 3 -> 32) ----
    {
        long long tot = (long long)B * 1024 * 32;
        k_transpose_pad<<<(unsigned)((tot + 255) / 256), 256, 0, stream>>>(
            x_in, xyzA, embIn, B, 1024);
        k_pad_w<<<(64 * 32 + 255) / 256, 256, 0, stream>>>(emb_w0, embW0p, 64);
    }
    run_gemm(stream, embIn, embW0p, nullptr, proj, B * 1024, 64, 32, 32, 32, 64,
             0, 0, 0, 1, 1, 0);
    run_bn(stream, proj, stats, emb_g0, emb_b0, nullptr, proj, (long long)B * 1024, 64, 1);
    run_gemm(stream, proj, emb_w1, nullptr, att, B * 1024, 64, 64, 64, 64, 64,
             0, 0, 0, 1, 1, 0);
    run_bn(stream, att, stats, emb_g1, emb_b1, nullptr, featsA, (long long)B * 1024, 64, 1);

    // ---- stages ----
    float* xyz = xyzA; float* nxyz = xyzB;
    float* feats = featsA; float* fnext = featsB;
    int Nin = 1024, din = 64;
    const int SARR[4] = {512, 256, 128, 64};

    for (int s = 0; s < 4; ++s) {
        const int S = SARR[s];
        const int dd = 2 * din;
        const int sbase = 12 + s * 40;
        BlockP pos0 = get_block(d_in, sbase + 0);
        BlockP pos1 = get_block(d_in, sbase + 10);
        BlockP pre0 = get_block(d_in, sbase + 20);
        BlockP pre1 = get_block(d_in, sbase + 30);

        k_fps<<<B, 256, 0, stream>>>(xyz, fpsIdx, Nin, S);
        {
            long long tg = (long long)B * S * 3;
            k_gather_xyz<<<(unsigned)((tg + 255) / 256), 256, 0, stream>>>(
                xyz, fpsIdx, nxyz, B, Nin, S);
        }
        k_knn<<<(B * S + 127) / 128, 128, 0, stream>>>(xyz, nxyz, knnIdx, B, Nin, S);
        {
            long long tgrp = (long long)B * S * 32 * din;
            k_group<<<(unsigned)((tgrp + 255) / 256), 256, 0, stream>>>(
                feats, fpsIdx, knnIdx, Y, B, Nin, S, din);
        }

        // pre blocks on [B*S groups, n=32, dd]
        run_block(stream, Y, (long long)B * S, 32, dd, pre0, q, kb, v, o, scores, proj, att, stats);
        run_block(stream, Y, (long long)B * S, 32, dd, pre1, q, kb, v, o, scores, proj, att, stats);

        // max-pool over K=32 -> [B*S, dd]
        {
            long long tp = (long long)B * S * dd;
            k_maxpool<<<(unsigned)((tp + 255) / 256), 256, 0, stream>>>(
                Y, fnext, (long long)B * S, 32, dd);
        }

        // pos blocks on [B groups, n=S, dd]
        run_block(stream, fnext, (long long)B, S, dd, pos0, q, kb, v, o, scores, proj, att, stats);
        run_block(stream, fnext, (long long)B, S, dd, pos1, q, kb, v, o, scores, proj, att, stats);

        float* t;
        t = xyz; xyz = nxyz; nxyz = t;
        t = feats; feats = fnext; fnext = t;
        Nin = S; din = dd;
    }

    // ---- classifier ----
    {
        long long tp = (long long)B * 1024;
        k_maxpool<<<(unsigned)((tp + 255) / 256), 256, 0, stream>>>(
            feats, pooled, (long long)B, 64, 1024);
    }
    run_gemm(stream, pooled, cls_w1, cls_b1, z1, B, 256, 1024, 1024, 1024, 256,
             0, 0, 0, 1, 1, 0);
    run_bn(stream, z1, stats, cls_g, cls_b, nullptr, z1, (long long)B, 256, 1);
    run_gemm(stream, z1, cls_w2, cls_b2, (float*)d_out, B, 40, 256, 256, 256, 40,
             0, 0, 0, 1, 1, 0);
}
